// LevelBiLSTM_41025527611706
// MI455X (gfx1250) — matrix operand order repaired and checked by reading, not compiled, not run on hardware
//
#include <hip/hip_runtime.h>

// ---------------------------------------------------------------------------
// Problem constants (from reference): B=1024, C=256, L=128, H=256, W=3, O=256
// ---------------------------------------------------------------------------
#define BB 1024
#define CC 256
#define LL 128
#define HH 256
#define WW 3
#define OO 256
#define GG 1024            // 4*H gate width
#define MROWS (BB * LL)    // 131072 rows (time-major: row = l*B + b)

typedef __attribute__((ext_vector_type(16))) __bf16 v16bf;
typedef __attribute__((ext_vector_type(8)))  float  v8f;

// f32 -> bf16 round-to-nearest-even without relying on __bf16 arithmetic
__device__ __forceinline__ __bf16 f2bf(float f) {
  unsigned u = __float_as_uint(f);
  unsigned r = (u + 0x7FFFu + ((u >> 16) & 1u)) >> 16;
  union { unsigned short s; __bf16 b; } cv;
  cv.s = (unsigned short)r;
  return cv.b;
}

// A-fragment (16-bit A 16x32 layout, ISA 7.12.2):
// lane<16 holds row m=lane, K = kbase+0..7 and kbase+16..23 (kbase = ks*32)
// lane>=16 holds row m=lane-16, same but kbase += 8.
__device__ __forceinline__ v16bf load_a16(const __bf16* rowbase, int kbase) {
  union { uint4 q[2]; v16bf v; } u;
  u.q[0] = *(const uint4*)(rowbase + kbase);
  u.q[1] = *(const uint4*)(rowbase + kbase + 16);
  return u.v;
}

__device__ __forceinline__ float sigmoidf_(float x) {
  return 1.0f / (1.0f + __expf(-x));
}

// Stage a contiguous block of bf16 data into LDS with the CDNA5 async path
// (GLOBAL_LOAD_ASYNC_TO_LDS_B128, ASYNCcnt-tracked; ISA 15.18.3 op 98).
// nch16 = number of 16-byte chunks. Follow with __syncthreads().
__device__ __forceinline__ void async_stage(__bf16* dstLds, const __bf16* srcG,
                                            int nch16, int tid) {
  for (int c = tid; c < nch16; c += 256) {
    unsigned ldsoff = (unsigned)(uintptr_t)dstLds + (unsigned)c * 16u;
    const char* g = (const char*)srcG + (size_t)c * 16;
    asm volatile("global_load_async_to_lds_b128 %0, %1, off"
                 :: "v"(ldsoff), "v"(g) : "memory");
  }
  asm volatile("s_wait_asynccnt 0x0" ::: "memory");
}

// ---------------------------------------------------------------------------
// Weight pre-pack into B-fragment order:
// Bp[((ntile*ksteps + kstep)*32 + lane)*16 + j] = W[kstep*32 + (j>>3)*16 + (lane>>4)*8 + (j&7)]
//                                                   [ntile*16 + (lane&15)]
// -> each wave B-fragment load is 32 lanes x 32B contiguous.
// ---------------------------------------------------------------------------
__global__ void pack_w_kernel(__bf16* __restrict__ dst, const float* __restrict__ src,
                              int K, int N, int ld) {
  int idx = blockIdx.x * 256 + threadIdx.x;
  if (idx >= K * N) return;
  int j     = idx & 15;
  int lane  = (idx >> 4) & 31;
  int t2    = idx >> 9;
  int ksteps = K >> 5;
  int kstep = t2 % ksteps;
  int ntile = t2 / ksteps;
  int k = kstep * 32 + ((j >> 3) << 4) + ((lane >> 4) << 3) + (j & 7);
  int n = ntile * 16 + (lane & 15);
  dst[idx] = f2bf(src[(size_t)k * ld + n]);
}

// Windowed layer-0 input, bf16, time-major rows: xw[(l*B+b)*768 + w*C + c]
__global__ void pack_xw_kernel(__bf16* __restrict__ dst, const float* __restrict__ x) {
  size_t idx = (size_t)blockIdx.x * 256 + threadIdx.x;
  if (idx >= (size_t)MROWS * (WW * CC)) return;
  size_t row = idx / (WW * CC);
  int k = (int)(idx % (WW * CC));
  int l = (int)(row >> 10);          // row = l*1024 + b
  int b = (int)(row & 1023);
  int w = k / CC, c = k % CC;
  int sl = l + w - 1;                // pad_start = 1 for W=3
  float v = (sl >= 0 && sl < LL) ? x[((size_t)b * CC + c) * LL + sl] : 0.0f;
  dst[idx] = f2bf(v);
}

// ---------------------------------------------------------------------------
// Input-projection GEMM: out = A(64 x K) @ W(K x 1024) + bias, per block.
// 8 waves; each wave owns 8 N-tiles x 4 M-tiles -> every B-fragment feeds
// 4 WMMAs (x4 L2-bandwidth amortization). A staged via async-to-LDS.
// gx output is a >0.5GB stream (>> 192MB L2): store non-temporal.
// Dynamic LDS: 64*K*2 bytes.
// ---------------------------------------------------------------------------
__global__ __launch_bounds__(256) void gemm_x_kernel(
    const __bf16* __restrict__ A, const __bf16* __restrict__ Bp,
    const float* __restrict__ bias, float* __restrict__ out, int K) {
  extern __shared__ __align__(16) char smem_gx[];
  __bf16* As = (__bf16*)smem_gx;             // 64 x K
  const int tid  = threadIdx.x;
  const int lane = tid & 31;
  const int wv   = tid >> 5;
  const size_t blk = blockIdx.x;
  const int ksteps = K >> 5;

  // 64 consecutive rows of A form one contiguous 64*K block.
  async_stage(As, A + blk * 64 * (size_t)K, 8 * K, tid);
  __syncthreads();

  v8f acc[8][4] = {};
  const int klaneoff = (lane >> 4) << 3;
  const size_t ntstride = (size_t)ksteps * 512;
  const __bf16* arows[4];
#pragma unroll
  for (int mt = 0; mt < 4; ++mt)
    arows[mt] = As + (size_t)(mt * 16 + (lane & 15)) * K;

  for (int ks = 0; ks < ksteps; ++ks) {
    v16bf a[4];
#pragma unroll
    for (int mt = 0; mt < 4; ++mt) a[mt] = load_a16(arows[mt], ks * 32 + klaneoff);
    const __bf16* bpp = Bp + ((size_t)(wv * 8) * ksteps + ks) * 512 + lane * 16;
    __builtin_prefetch(bpp + 512, 0, 1);     // next kstep chunk -> global_prefetch
#pragma unroll
    for (int nt = 0; nt < 8; ++nt) {
      v16bf b = *(const v16bf*)bpp;
#pragma unroll
      for (int mt = 0; mt < 4; ++mt)
        acc[nt][mt] = __builtin_amdgcn_wmma_f32_16x16x32_bf16(
            false, a[mt], false, b, (short)0, acc[nt][mt], false, false);
      bpp += ntstride;
    }
  }

  const int mb2 = (lane >> 4) << 3;          // C/D layout: m = mb2 + v, n = lane&15
#pragma unroll
  for (int nt = 0; nt < 8; ++nt) {
    int n0 = (wv * 8 + nt) * 16 + (lane & 15);
    float bv = bias[n0];
#pragma unroll
    for (int mt = 0; mt < 4; ++mt)
#pragma unroll
      for (int v = 0; v < 8; ++v)
        __builtin_nontemporal_store(
            acc[nt][mt][v] + bv,
            &out[(blk * 64 + mt * 16 + mb2 + v) * (size_t)GG + n0]);
  }
}

// ---------------------------------------------------------------------------
// Persistent fused bi-LSTM layer: grid (B/32, 2 dirs); each workgroup owns 32
// batch rows for the ENTIRE time loop (recurrence is batch-parallel, so no
// cross-workgroup sync is needed). h lives in a 16KB LDS tile (WMMA A source
// via ds_load_b128), c lives in registers, gates staged in a 128KB LDS tile
// for the cross-wave cell math. Each Wh B-fragment feeds 2 WMMAs.
// Wh stays L2-resident; gx is streamed NT.
// Dynamic LDS: 32*1024*4 (gates) + 32*256*2 (h) = 147456 bytes.
// ---------------------------------------------------------------------------
__global__ __launch_bounds__(256) void lstm_layer_kernel(
    const __bf16* __restrict__ Wh0, const __bf16* __restrict__ Wh1,
    const float* __restrict__ gx0, const float* __restrict__ gx1,
    __bf16* __restrict__ xsout) {
  extern __shared__ __align__(16) char smem[];
  float*  gates = (float*)smem;                       // 32 x 1024 f32
  __bf16* hs    = (__bf16*)(smem + 32 * GG * 4);      // 32 x 256 bf16

  const int tid  = threadIdx.x;
  const int lane = tid & 31;
  const int wv   = tid >> 5;
  const int mb   = blockIdx.x;
  const int dir  = blockIdx.y;
  const __bf16* Wh = dir ? Wh1 : Wh0;
  const float*  gx = dir ? gx1 : gx0;
  const int b0 = mb * 32;

  // init h = 0 (LDS), c = 0 (registers; fixed idx = it*256+tid mapping)
  float creg[32];
#pragma unroll
  for (int it = 0; it < 32; ++it) creg[it] = 0.0f;
  for (int i = tid; i < 32 * HH; i += 256) hs[i] = f2bf(0.0f);
  __syncthreads();

  const __bf16* arows[2];
#pragma unroll
  for (int mt = 0; mt < 2; ++mt)
    arows[mt] = hs + (size_t)(mt * 16 + (lane & 15)) * HH;
  const int klaneoff = (lane >> 4) << 3;
  const int mb2 = (lane >> 4) << 3;
  const size_t ntstride = (size_t)8 * 512;            // ksteps = HH/32 = 8

  for (int t = 0; t < LL; ++t) {
    const int l = dir ? (LL - 1 - t) : t;

    // ---- recurrent GEMM: gates_h = h_prev @ Wh (32x1024 tile) ----
    v8f acc[8][2] = {};
#pragma unroll
    for (int ks = 0; ks < 8; ++ks) {
      v16bf a[2];
#pragma unroll
      for (int mt = 0; mt < 2; ++mt) a[mt] = load_a16(arows[mt], ks * 32 + klaneoff);
      const __bf16* bpp = Wh + ((size_t)(wv * 8) * 8 + ks) * 512 + lane * 16;
      __builtin_prefetch(bpp + 512, 0, 1);
#pragma unroll
      for (int nt = 0; nt < 8; ++nt) {
        v16bf b = *(const v16bf*)bpp;
#pragma unroll
        for (int mt = 0; mt < 2; ++mt)
          acc[nt][mt] = __builtin_amdgcn_wmma_f32_16x16x32_bf16(
              false, a[mt], false, b, (short)0, acc[nt][mt], false, false);
        bpp += ntstride;
      }
    }

    // ---- gates = gates_h + gx[l] (NT stream read), staged to LDS ----
    const size_t growbase = ((size_t)l * BB + b0) * GG;
#pragma unroll
    for (int nt = 0; nt < 8; ++nt) {
      int n0 = (wv * 8 + nt) * 16 + (lane & 15);
#pragma unroll
      for (int mt = 0; mt < 2; ++mt)
#pragma unroll
        for (int v = 0; v < 8; ++v) {
          int m = mt * 16 + mb2 + v;
          gates[m * GG + n0] =
              acc[nt][mt][v] +
              __builtin_nontemporal_load(&gx[growbase + (size_t)m * GG + n0]);
        }
    }
    __syncthreads();

    // ---- LSTM cell: 32 rows x 256 H, 32 elems/thread, c in registers ----
#pragma unroll
    for (int it = 0; it < 32; ++it) {
      int idx = it * 256 + tid;
      int m  = idx >> 8;
      int hc = idx & 255;
      float gi = gates[m * GG + hc];
      float gg = gates[m * GG + 256 + hc];
      float gf = gates[m * GG + 512 + hc];
      float go = gates[m * GG + 768 + hc];
      float cn = sigmoidf_(gf + 1.0f) * creg[it] + sigmoidf_(gi) * tanhf(gg);
      float hn = sigmoidf_(go) * tanhf(cn);
      creg[it] = cn;
      __bf16 hb = f2bf(hn);
      hs[m * HH + hc] = hb;
      xsout[((size_t)l * BB + b0 + m) * (2 * HH) + dir * HH + hc] = hb;
    }
    __syncthreads();   // hs/gates stable before next step's GEMM
  }
}

// ---------------------------------------------------------------------------
// Output projection: (L*B, 512) @ (512, 256) + b_out, stored as (B, O, L).
// M=64 per block; 8 waves x (2 N-tiles x 4 M-tiles). Dynamic LDS: 64KB.
// ---------------------------------------------------------------------------
__global__ __launch_bounds__(256) void gemm_out_kernel(
    const __bf16* __restrict__ A, const __bf16* __restrict__ Bp,
    const float* __restrict__ bias, float* __restrict__ dout) {
  extern __shared__ __align__(16) char smem_go[];
  __bf16* As = (__bf16*)smem_go;             // 64 x 512
  const int tid  = threadIdx.x;
  const int lane = tid & 31;
  const int wv   = tid >> 5;
  const size_t blk = blockIdx.x;
  const int K = 512, ksteps = 16;

  async_stage(As, A + blk * 64 * (size_t)K, 8 * K, tid);
  __syncthreads();

  v8f acc[2][4] = {};
  const int klaneoff = (lane >> 4) << 3;
  const size_t ntstride = (size_t)ksteps * 512;
  const __bf16* arows[4];
#pragma unroll
  for (int mt = 0; mt < 4; ++mt)
    arows[mt] = As + (size_t)(mt * 16 + (lane & 15)) * K;

  for (int ks = 0; ks < ksteps; ++ks) {
    v16bf a[4];
#pragma unroll
    for (int mt = 0; mt < 4; ++mt) a[mt] = load_a16(arows[mt], ks * 32 + klaneoff);
    const __bf16* bpp = Bp + ((size_t)(wv * 2) * ksteps + ks) * 512 + lane * 16;
#pragma unroll
    for (int nt = 0; nt < 2; ++nt) {
      v16bf b = *(const v16bf*)bpp;
#pragma unroll
      for (int mt = 0; mt < 4; ++mt)
        acc[nt][mt] = __builtin_amdgcn_wmma_f32_16x16x32_bf16(
            false, a[mt], false, b, (short)0, acc[nt][mt], false, false);
      bpp += ntstride;
    }
  }

  const int mb2 = (lane >> 4) << 3;
  const int l  = (int)(blk >> 4);            // row0/1024, row0 = blk*64
  const int b0 = ((int)blk & 15) * 64;       // row0%1024
#pragma unroll
  for (int nt = 0; nt < 2; ++nt) {
    int n0 = (wv * 2 + nt) * 16 + (lane & 15);
    float bv = bias[n0];
#pragma unroll
    for (int mt = 0; mt < 4; ++mt)
#pragma unroll
      for (int v = 0; v < 8; ++v)
        __builtin_nontemporal_store(
            acc[nt][mt][v] + bv,
            &dout[((size_t)(b0 + mt * 16 + mb2 + v) * OO + n0) * LL + l]);
  }
}

// ---------------------------------------------------------------------------
extern "C" void kernel_launch(void* const* d_in, const int* in_sizes, int n_in,
                              void* d_out, int out_size, void* d_ws, size_t ws_size,
                              hipStream_t stream) {
  (void)in_sizes; (void)n_in; (void)out_size; (void)ws_size;
  const float* inputs = (const float*)d_in[0];
  const float* w_fw0  = (const float*)d_in[1];
  const float* b_fw0  = (const float*)d_in[2];
  const float* w_bw0  = (const float*)d_in[3];
  const float* b_bw0  = (const float*)d_in[4];
  const float* w_fw_r = (const float*)d_in[5];
  const float* b_fw_r = (const float*)d_in[6];
  const float* w_bw_r = (const float*)d_in[7];
  const float* b_bw_r = (const float*)d_in[8];
  const float* w_out  = (const float*)d_in[9];
  const float* b_out  = (const float*)d_in[10];
  float* out = (float*)d_out;

  char* ws = (char*)d_ws;
  size_t off = 0;
  auto alloc = [&](size_t bytes) -> char* {
    char* p = ws + off;
    off += (bytes + 255) & ~(size_t)255;
    return p;
  };

  __bf16* Wx0[2];   // layer0 x-projection weights (K=768)
  __bf16* Wh[3][2]; // per-layer recurrent weights (K=256)
  __bf16* Wxr[2][2];// layers 1,2 x-projection weights (K=512)
  Wx0[0] = (__bf16*)alloc((size_t)768 * 1024 * 2);
  Wx0[1] = (__bf16*)alloc((size_t)768 * 1024 * 2);
  for (int L2 = 0; L2 < 3; ++L2)
    for (int d = 0; d < 2; ++d) Wh[L2][d] = (__bf16*)alloc((size_t)256 * 1024 * 2);
  for (int k = 0; k < 2; ++k)
    for (int d = 0; d < 2; ++d) Wxr[k][d] = (__bf16*)alloc((size_t)512 * 1024 * 2);
  __bf16* Woutp = (__bf16*)alloc((size_t)512 * 256 * 2);

  __bf16* xw  = (__bf16*)alloc((size_t)MROWS * 768 * 2);
  __bf16* xsA = (__bf16*)alloc((size_t)MROWS * 512 * 2);
  __bf16* xsB = (__bf16*)alloc((size_t)MROWS * 512 * 2);
  float* gx0  = (float*)alloc((size_t)MROWS * GG * 4);
  float* gx1  = (float*)alloc((size_t)MROWS * GG * 4);

  auto packw = [&](__bf16* dstp, const float* srcp, int K, int N, int ld) {
    int total = K * N;
    pack_w_kernel<<<(total + 255) / 256, 256, 0, stream>>>(dstp, srcp, K, N, ld);
  };

  // Split each weight into x-part (rows [0..K_x)) and h-part (last 256 rows).
  packw(Wx0[0],   w_fw0,                         768, 1024, 1024);
  packw(Wh[0][0], w_fw0 + (size_t)768 * 1024,    256, 1024, 1024);
  packw(Wx0[1],   w_bw0,                         768, 1024, 1024);
  packw(Wh[0][1], w_bw0 + (size_t)768 * 1024,    256, 1024, 1024);
  for (int k = 0; k < 2; ++k) {
    const float* wf = w_fw_r + (size_t)k * 768 * 1024;
    const float* wb = w_bw_r + (size_t)k * 768 * 1024;
    packw(Wxr[k][0],    wf,                       512, 1024, 1024);
    packw(Wh[k + 1][0], wf + (size_t)512 * 1024,  256, 1024, 1024);
    packw(Wxr[k][1],    wb,                       512, 1024, 1024);
    packw(Wh[k + 1][1], wb + (size_t)512 * 1024,  256, 1024, 1024);
  }
  packw(Woutp, w_out, 512, 256, 256);

  {
    size_t total = (size_t)MROWS * (WW * CC);
    pack_xw_kernel<<<(unsigned)((total + 255) / 256), 256, 0, stream>>>(xw, inputs);
  }

  const __bf16* layer_in = xw;
  int Kin = 768;
  __bf16* xs_cur = xsA;
  __bf16* xs_nxt = xsB;
  const unsigned layer_smem = 32 * GG * 4 + 32 * HH * 2;   // 147456 B

  for (int layer = 0; layer < 3; ++layer) {
    const __bf16 *wx_f, *wx_b;
    const float *bias_f, *bias_b;
    if (layer == 0) {
      wx_f = Wx0[0]; wx_b = Wx0[1]; bias_f = b_fw0; bias_b = b_bw0;
    } else {
      wx_f = Wxr[layer - 1][0]; wx_b = Wxr[layer - 1][1];
      bias_f = b_fw_r + (size_t)(layer - 1) * GG;
      bias_b = b_bw_r + (size_t)(layer - 1) * GG;
    }

    const unsigned gx_smem = 64u * (unsigned)Kin * 2u;
    gemm_x_kernel<<<MROWS / 64, 256, gx_smem, stream>>>(layer_in, wx_f, bias_f, gx0, Kin);
    gemm_x_kernel<<<MROWS / 64, 256, gx_smem, stream>>>(layer_in, wx_b, bias_b, gx1, Kin);

    // One persistent launch replaces 128 per-step launches (both directions).
    lstm_layer_kernel<<<dim3(BB / 32, 2), 256, layer_smem, stream>>>(
        Wh[layer][0], Wh[layer][1], gx0, gx1, xs_cur);

    layer_in = xs_cur;
    Kin = 512;
    __bf16* tmp = xs_cur; xs_cur = xs_nxt; xs_nxt = tmp;
  }

  gemm_out_kernel<<<MROWS / 64, 256, 64 * 512 * 2, stream>>>(layer_in, Woutp, b_out, out);
}
